// KoLeoLoss_26680336842882
// MI455X (gfx1250) — matrix operand order repaired and checked by reading, not compiled, not run on hardware
//
#include <hip/hip_runtime.h>
#include <hip/hip_bf16.h>

typedef __attribute__((ext_vector_type(16))) _Float16 v16h;
typedef __attribute__((ext_vector_type(8)))  _Float16 v8h;
typedef __attribute__((ext_vector_type(8)))  float    v8f;

#define N_ROWS 8192
#define DIM    1024
#define NSEG   8              // column range split for occupancy
#define EPSV   1e-8f
#define LDS_STRIDE 1032       // 1024 + 8 halves pad -> conflict-free ds_load_b128
#define LDS_BYTES  (64 * LDS_STRIDE * 2)   // 132096 B, fits 2 WGs in 320KB WGP LDS

// ---------------------------------------------------------------------------
// Kernel 1: row L2 normalize -> fp32 copy (exact distances) + f16 copy (WMMA).
// ---------------------------------------------------------------------------
__global__ void koleo_normalize(const float* __restrict__ x,
                                float* __restrict__ xn,
                                _Float16* __restrict__ xh) {
    const int lane = threadIdx.x & 31;
    const int wave = threadIdx.x >> 5;
    const int row  = blockIdx.x * 8 + wave;
    const float* src = x + (size_t)row * DIM;

    float s = 0.f;
#pragma unroll
    for (int d = lane; d < DIM; d += 32) { float v = src[d]; s += v * v; }
#pragma unroll
    for (int m = 1; m < 32; m <<= 1) s += __shfl_xor(s, m, 32);

    const float scale = 1.0f / fmaxf(sqrtf(s), EPSV);

    float*    drow = xn + (size_t)row * DIM;
    _Float16* hrow = xh + (size_t)row * DIM;
#pragma unroll
    for (int d = lane; d < DIM; d += 32) {
        float v = src[d] * scale;
        drow[d] = v;
        hrow[d] = (_Float16)v;
    }
}

// ---------------------------------------------------------------------------
// Kernel 2: fused sim = Xn*Xn^T + diag mask + running argmax.
// Wave tile: 32 rows x 64 cols (2 A frags x 4 B frags, 8 f32 accumulators).
// B K-panel (64 cols x 1024 K, 128KB) staged in LDS, shared by 8 waves.
// Per-tile loop-invariant LDS base pointers keep every ds_load offset inside
// the 16-bit immediate (no in-loop VALU address math), and B fragments are
// ping-pong double-buffered so no ds_load writes an octet inside the 4-slot
// WMMA WAR hazard window (kills the v_nop blocks).
// ---------------------------------------------------------------------------
__global__ void koleo_sim_argmax(const _Float16* __restrict__ xh,
                                 float* __restrict__ pval,
                                 int*   __restrict__ pidx) {
    extern __shared__ _Float16 ldsB[];   // [64][LDS_STRIDE]

    const int lane  = threadIdx.x & 31;
    const int wave  = threadIdx.x >> 5;
    const int half  = lane >> 4;         // 0: lanes 0-15, 1: lanes 16-31
    const int l15   = lane & 15;
    const int mBase = blockIdx.x * 256 + wave * 32;
    const int segBase = blockIdx.y * (N_ROWS / NSEG);   // 1024 cols / segment

    // A fragment row pointers (lane l15 holds matrix row M = l15, both halves)
    const _Float16* arow0 = xh + (size_t)(mBase + l15) * DIM;
    const _Float16* arow1 = xh + (size_t)(mBase + 16 + l15) * DIM;

    // loop-invariant per-tile LDS base pointers (lane = N column within tile)
    const _Float16* bl[4];
#pragma unroll
    for (int t = 0; t < 4; ++t)
        bl[t] = ldsB + (t * 16 + l15) * LDS_STRIDE + half * 16;

    float bestv[2][8];
    int   besti[2][8];
#pragma unroll
    for (int i = 0; i < 2; ++i)
#pragma unroll
        for (int r = 0; r < 8; ++r) { bestv[i][r] = -1e30f; besti[i][r] = 0; }

    for (int g = 0; g < 16; ++g) {                 // 16 groups of 64 cols
        const int colBase0 = segBase + g * 64;

        // ---- cooperative stage of B K-panel into LDS (64 rows x 2KB) ----
        __syncthreads();                           // previous group's reads done
        for (int ch = threadIdx.x; ch < 64 * 128; ch += 256) {
            const int r = ch >> 7;                 // col row 0..63
            const int o = (ch & 127) * 8;          // halves within row
            v8h d = *(const v8h*)(xh + (size_t)(colBase0 + r) * DIM + o);
            *(v8h*)(&ldsB[r * LDS_STRIDE + o]) = d;
        }
        __syncthreads();

        v8f c[2][4];
#pragma unroll
        for (int i = 0; i < 2; ++i)
#pragma unroll
            for (int t = 0; t < 4; ++t)
                c[i][t] = v8f{0.f, 0.f, 0.f, 0.f, 0.f, 0.f, 0.f, 0.f};

        // one half-step: prefetch B[kpre] into bpre, A[kb] from global,
        // then the 8-WMMA burst on buse.
        auto halfstep = [&](int kb, int kpre, v16h* buse, v16h* bpre) {
#pragma unroll
            for (int t = 0; t < 4; ++t)
                bpre[t] = *(const v16h*)(bl[t] + kpre);

            v8h a0lo = *(const v8h*)(arow0 + kb + half * 16);
            v8h a0hi = *(const v8h*)(arow0 + kb + 8 + half * 16);
            v8h a1lo = *(const v8h*)(arow1 + kb + half * 16);
            v8h a1hi = *(const v8h*)(arow1 + kb + 8 + half * 16);
            v16h a0, a1;
#pragma unroll
            for (int i = 0; i < 8; ++i) {
                a0[i] = a0lo[i]; a0[i + 8] = a0hi[i];
                a1[i] = a1lo[i]; a1[i + 8] = a1hi[i];
            }

#pragma unroll
            for (int t = 0; t < 4; ++t) {
                c[0][t] = __builtin_amdgcn_wmma_f32_16x16x32_f16(
                              false, a0, false, buse[t], (short)0, c[0][t], false, false);
                c[1][t] = __builtin_amdgcn_wmma_f32_16x16x32_f16(
                              false, a1, false, buse[t], (short)0, c[1][t], false, false);
            }
        };

        v16h bA[4], bB[4];
#pragma unroll
        for (int t = 0; t < 4; ++t)                 // prologue: B[0] -> bA
            bA[t] = *(const v16h*)(bl[t]);

        for (int kb = 0; kb < DIM; kb += 64) {
            // use bA for kb, prefetch kb+32 into bB
            halfstep(kb, kb + 32, bA, bB);
            // use bB for kb+32, prefetch (kb+64) mod DIM into bA
            // (wrapped dummy load on the last step keeps this branch-free)
            halfstep(kb + 32, (kb + 64) & (DIM - 1), bB, bA);
        }

        // C layout: VGPR r -> row (base + half*8 + r), col = colBase + l15
#pragma unroll
        for (int i = 0; i < 2; ++i)
#pragma unroll
            for (int t = 0; t < 4; ++t) {
                const int col = colBase0 + t * 16 + l15;
#pragma unroll
                for (int r = 0; r < 8; ++r) {
                    const int row = mBase + i * 16 + half * 8 + r;
                    float v = c[i][t][r];
                    if (col == row) v = -1.0f;     // mask self-similarity
                    if (v > bestv[i][r]) { bestv[i][r] = v; besti[i][r] = col; }
                }
            }
    }

    // Reduce argmax across the 16 lanes sharing each row (ties -> lower index)
#pragma unroll
    for (int i = 0; i < 2; ++i)
#pragma unroll
        for (int r = 0; r < 8; ++r) {
            float v = bestv[i][r];
            int   j = besti[i][r];
#pragma unroll
            for (int m = 1; m < 16; m <<= 1) {
                float ov = __shfl_xor(v, m, 32);
                int   oj = __shfl_xor(j, m, 32);
                if (ov > v || (ov == v && oj < j)) { v = ov; j = oj; }
            }
            if (l15 == 0) {
                const int row = mBase + i * 16 + half * 8 + r;
                pval[blockIdx.y * N_ROWS + row] = v;
                pidx[blockIdx.y * N_ROWS + row] = j;
            }
        }
}

// ---------------------------------------------------------------------------
// Kernel 2b: combine the NSEG column-segment partial argmaxes per row.
// ---------------------------------------------------------------------------
__global__ void koleo_argmax_combine(const float* __restrict__ pval,
                                     const int* __restrict__ pidx,
                                     int* __restrict__ idx) {
    const int row = blockIdx.x * blockDim.x + threadIdx.x;
    if (row >= N_ROWS) return;
    float bv = -1e30f;
    int   bi = 0;
    for (int s = 0; s < NSEG; ++s) {        // ascending -> first max wins ties
        float v = pval[s * N_ROWS + row];
        if (v > bv) { bv = v; bi = pidx[s * N_ROWS + row]; }
    }
    idx[row] = bi;
}

// ---------------------------------------------------------------------------
// Kernel 3: pairwise distance in fp32 + -mean(log(dist+eps)). One wave/row.
// ---------------------------------------------------------------------------
__global__ void koleo_loss(const float* __restrict__ xn,
                           const int* __restrict__ idx,
                           float* __restrict__ out) {
    const int lane = threadIdx.x & 31;
    const int wave = threadIdx.x >> 5;
    const int row  = blockIdx.x * 8 + wave;
    const float* a = xn + (size_t)row * DIM;
    const float* b = xn + (size_t)idx[row] * DIM;

    float s = 0.f;
#pragma unroll
    for (int d = lane; d < DIM; d += 32) {
        float df = a[d] - b[d] + EPSV;
        s += df * df;
    }
#pragma unroll
    for (int m = 1; m < 32; m <<= 1) s += __shfl_xor(s, m, 32);

    if (lane == 0) {
        float dist = sqrtf(s);
        atomicAdd(out, -logf(dist + EPSV) * (1.0f / (float)N_ROWS));
    }
}

// ---------------------------------------------------------------------------
extern "C" void kernel_launch(void* const* d_in, const int* in_sizes, int n_in,
                              void* d_out, int out_size, void* d_ws, size_t ws_size,
                              hipStream_t stream) {
    const float* x   = (const float*)d_in[0];
    float*       out = (float*)d_out;

    char* ws = (char*)d_ws;
    float*    xn   = (float*)ws;                                       // 32 MB fp32 normalized
    _Float16* xh   = (_Float16*)(ws + (size_t)32 * 1024 * 1024);       // 16 MB f16 normalized
    float*    pval = (float*)(ws + (size_t)48 * 1024 * 1024);          // 256 KB partial max vals
    int*      pidx = (int*)  (ws + (size_t)48 * 1024 * 1024 + 256 * 1024); // 256 KB partial idx
    int*      idx  = (int*)  (ws + (size_t)48 * 1024 * 1024 + 512 * 1024); // 32 KB final idx

    // allow >64KB dynamic LDS for the B-panel staging kernel
    (void)hipFuncSetAttribute((const void*)koleo_sim_argmax,
                              hipFuncAttributeMaxDynamicSharedMemorySize,
                              LDS_BYTES);

    hipMemsetAsync(d_out, 0, sizeof(float), stream);

    koleo_normalize<<<N_ROWS / 8, 256, 0, stream>>>(x, xn, xh);

    dim3 grid2(N_ROWS / 256, NSEG);       // 32 row-blocks x 8 column segments
    koleo_sim_argmax<<<grid2, 256, LDS_BYTES, stream>>>(xh, pval, pidx);

    koleo_argmax_combine<<<N_ROWS / 256, 256, 0, stream>>>(pval, pidx, idx);

    koleo_loss<<<N_ROWS / 8, 256, 0, stream>>>(xn, idx, out);
}
